// Decoder_9887014716020
// MI455X (gfx1250) — compile-verified, matrix operand order checked by reading, not run on hardware
//
#include <hip/hip_runtime.h>
#include <math.h>

// Problem constants (B=2, H=W=64, HD=40, 6 nodes, 10 edges)
#define HWPIX 4096
#define PBITS 12
#define IMGW  64
#define MTOT  8192   // B * H * W

typedef __attribute__((ext_vector_type(16))) __bf16 v16bf;
typedef __attribute__((ext_vector_type(8)))  float  v8f;
typedef __attribute__((ext_vector_type(4)))  unsigned int uint4v;

union V16U { uint4v q[2]; v16bf v; };

// gfx1250 async global->LDS staging (fallback to sync path if the toolchain
// doesn't expose the builtins)
#if defined(__has_builtin)
#if __has_builtin(__builtin_amdgcn_global_load_async_to_lds_b128) && \
    __has_builtin(__builtin_amdgcn_s_wait_asynccnt)
#define GEMM_ASYNC_LDS 1
#endif
#endif

#ifdef GEMM_ASYNC_LDS
// Builtin signature (from clang diagnostic): param0 is
// 'int __attribute__((vector_size(16))) __device__ *' (addrspace(1), non-const)
typedef int v4i_vs __attribute__((vector_size(16)));
typedef __attribute__((address_space(1))) v4i_vs* gas_ptr;   // __device__
typedef __attribute__((address_space(3))) v4i_vs* las_ptr;   // __shared__
#endif

__device__ __forceinline__ unsigned short f2bf(float f) {
  unsigned int u = __float_as_uint(f);
  u = u + 0x7FFFu + ((u >> 16) & 1u);   // round-to-nearest-even
  return (unsigned short)(u >> 16);
}
__device__ __forceinline__ float sigm(float x) { return 1.0f / (1.0f + __expf(-x)); }

// ---------------------------------------------------------------------------
// BF16 WMMA GEMM: out[m,n] = epi( sum_k A[m,k] * W[n,k] ), fp32 accumulate.
//  A: (M, lda) row-major bf16, K-loop length KLOOP (compile-time, mult of 32,
//     zero-padded by producers -> no K guards).
//  W: (NPAD, wlda) row-major bf16, rows AND columns zero-padded -> no N guards
//     in the loop (gridDim.y * 32 == NPAD).
//  out: outF!=null -> fp32 channel-major (B, N, HW), batch stride ostr
//       else       -> bf16 row-major (M, ostr)
// Tile 64(M) x 32(N), 4 waves; one A fragment feeds two WMMA chains.
// ---------------------------------------------------------------------------
template<int KLOOP>
__global__ __launch_bounds__(128)
void gemm_bf16_kernel(const unsigned short* __restrict__ A, long lda,
                      const unsigned short* __restrict__ W, long wlda,
                      const float* __restrict__ gamma, const float* __restrict__ beta,
                      float* __restrict__ outF, unsigned short* __restrict__ outB,
                      long ostr, int N, int relu)
{
  __shared__ unsigned short As[64][40];   // 80B rows: 16B aligned, conflict-free
  __shared__ unsigned short Bs[32][40];

  const int tid  = threadIdx.x;
  const int lane = tid & 31;
  const int wave = tid >> 5;
  const int m0   = blockIdx.x * 64;
  const int n0   = blockIdx.y * 32;
  const int kh   = lane >> 4;      // lane half -> K group (ISA 16-bit layout)
  const int row  = lane & 15;

  v8f acc0 = {0.f,0.f,0.f,0.f,0.f,0.f,0.f,0.f};
  v8f acc1 = {0.f,0.f,0.f,0.f,0.f,0.f,0.f,0.f};

  // A staging: 256 chunks of 8 bf16; thread t takes chunks t and t+128
  const int amm0 = tid >> 2,         akk = (tid & 3) * 8;
  const int amm1 = (tid + 128) >> 2;
  // B staging: 128 chunks (weights pre-padded: no guard)
  const int bnn  = tid >> 2,         bkk = akk;

  const unsigned short* gA0 = &A[(long)(m0 + amm0) * lda + akk];
  const unsigned short* gA1 = &A[(long)(m0 + amm1) * lda + akk];
  const unsigned short* gB  = &W[(long)(n0 + bnn) * wlda + bkk];

  for (int k0 = 0; k0 < KLOOP; k0 += 32) {
#ifdef GEMM_ASYNC_LDS
    __builtin_amdgcn_global_load_async_to_lds_b128(
        (gas_ptr)(unsigned short*)(gA0 + k0), (las_ptr)&As[amm0][akk], 0, 0);
    __builtin_amdgcn_global_load_async_to_lds_b128(
        (gas_ptr)(unsigned short*)(gA1 + k0), (las_ptr)&As[amm1][akk], 0, 0);
    __builtin_amdgcn_global_load_async_to_lds_b128(
        (gas_ptr)(unsigned short*)(gB + k0),  (las_ptr)&Bs[bnn][bkk], 0, 0);
    __builtin_amdgcn_s_wait_asynccnt(0);
#else
    uint4v a0 = *(const uint4v*)(gA0 + k0);
    uint4v a1 = *(const uint4v*)(gA1 + k0);
    uint4v bw = *(const uint4v*)(gB + k0);
    *(uint4v*)&As[amm0][akk] = a0;
    *(uint4v*)&As[amm1][akk] = a1;
    *(uint4v*)&Bs[bnn][bkk]  = bw;
#endif
    __syncthreads();

    V16U fa, fb0, fb1;
    fa.q[0]  = *(const uint4v*)&As[wave * 16 + row][kh * 8];
    fa.q[1]  = *(const uint4v*)&As[wave * 16 + row][16 + kh * 8];
    fb0.q[0] = *(const uint4v*)&Bs[row][kh * 8];
    fb0.q[1] = *(const uint4v*)&Bs[row][16 + kh * 8];
    fb1.q[0] = *(const uint4v*)&Bs[16 + row][kh * 8];
    fb1.q[1] = *(const uint4v*)&Bs[16 + row][16 + kh * 8];
    acc0 = __builtin_amdgcn_wmma_f32_16x16x32_bf16(
        false, fa.v, false, fb0.v, (short)0, acc0, false, false);
    acc1 = __builtin_amdgcn_wmma_f32_16x16x32_bf16(
        false, fa.v, false, fb1.v, (short)0, acc1, false, false);
    __syncthreads();
  }

  // C/D layout: lane half -> M+8, VGPR r -> row; lane&15 -> column
  const int nl = lane & 15;
  const int mb = m0 + wave * 16 + (lane >> 4) * 8;
#pragma unroll
  for (int t = 0; t < 2; ++t) {
    int n = n0 + t * 16 + nl;
    v8f acc = t ? acc1 : acc0;
    if (n < N) {
      float g  = gamma ? gamma[n] : 1.0f;
      float bb = beta  ? beta[n]  : 0.0f;
#pragma unroll
      for (int r = 0; r < 8; ++r) {
        int m = mb + r;
        float v = acc[r] * g + bb;
        if (relu) v = fmaxf(v, 0.0f);
        if (outF) {
          int b = m >> PBITS, p = m & (HWPIX - 1);
          outF[(long)b * ostr + (long)n * HWPIX + p] = v;
        } else {
          outB[(long)m * ostr + n] = f2bf(v);
        }
      }
    }
  }
}

// ---------------------------------------------------------------------------
// Weight convert+pad: fp32 groups of (NGRP, KR) -> bf16 groups of (NPAD, KP),
// zero padding both the K tail and the extra N rows (=> guard-free GEMM).
// ---------------------------------------------------------------------------
__global__ void wconv_kernel(const float* __restrict__ src, unsigned short* __restrict__ dst,
                             int NGRP, int NPAD, int KR, int KP, int total)
{
  int idx = blockIdx.x * blockDim.x + threadIdx.x;
  if (idx >= total) return;
  int nd = idx / KP, k = idx % KP;
  int g = nd / NPAD, r = nd % NPAD;
  float v = 0.0f;
  if (r < NGRP && k < KR) v = src[((long)g * NGRP + r) * KR + k];
  dst[idx] = f2bf(v);
}

// channel-major fp32 -> row-major bf16 pack (into column offset c0 of dst)
__global__ void pack_rows_kernel(const float* __restrict__ src, long sbst,
                                 unsigned short* __restrict__ dst, long dl,
                                 int c0, int C, int total /* MTOT*C */)
{
  int idx = blockIdx.x * blockDim.x + threadIdx.x;
  if (idx >= total) return;
  int c = idx % C, m = idx / C;
  int b = m >> PBITS, p = m & (HWPIX - 1);
  dst[(long)m * dl + c0 + c] = f2bf(src[(long)b * sbst + (long)c * HWPIX + p]);
}

// zero a bf16 column range [c0, c0+CZ) of a (M, dl) row-major buffer
__global__ void zero_cols_kernel(unsigned short* __restrict__ dst, long dl,
                                 int c0, int CZ, int total /* MTOT*CZ */)
{
  int idx = blockIdx.x * blockDim.x + threadIdx.x;
  if (idx >= total) return;
  int c = idx % CZ, m = idx / CZ;
  dst[(long)m * dl + c0 + c] = 0;
}

// ---------------------------------------------------------------------------
// im2col (3x3 SAME) over virtual concat (xA|xB), bf16 out, row stride 736
// ---------------------------------------------------------------------------
__global__ void im2col_kernel(const float* __restrict__ xA, const float* __restrict__ xB,
                              long bsA, long bsB, unsigned short* __restrict__ sbuf,
                              int C, int Chalf, int M)
{
  int idx = blockIdx.x * blockDim.x + threadIdx.x;
  if (idx >= M * C) return;
  int c = idx % C, m = idx / C;
  int b = m >> PBITS, p = m & (HWPIX - 1);
  int y = p >> 6, x = p & (IMGW - 1);
  const float* src = (c < Chalf) ? (xA + (long)b * bsA + (long)c * HWPIX)
                                 : (xB + (long)b * bsB + (long)(c - Chalf) * HWPIX);
  unsigned short* dst = sbuf + (long)m * 736 + c * 9;
#pragma unroll
  for (int tap = 0; tap < 9; ++tap) {
    int yy = y + tap / 3 - 1, xx = x + tap % 3 - 1;
    float v = (yy >= 0 && yy < IMGW && xx >= 0 && xx < IMGW) ? src[yy * IMGW + xx] : 0.0f;
    dst[tap] = f2bf(v);
  }
}

// ---------------------------------------------------------------------------
// Deformable bilinear sampling (reference clip + validity-masked weights,
// then sigmoid-mask scale), bf16 out, row stride 736
// ---------------------------------------------------------------------------
__global__ void dcn_sample_kernel(const float* __restrict__ xA, const float* __restrict__ xB,
                                  long bsA, long bsB, const float* __restrict__ om,
                                  unsigned short* __restrict__ sbuf,
                                  int C, int Chalf, int M)
{
  int idx = blockIdx.x * blockDim.x + threadIdx.x;
  if (idx >= M * C) return;
  int c = idx % C, m = idx / C;
  int b = m >> PBITS, p = m & (HWPIX - 1);
  int y = p >> 6, x = p & (IMGW - 1);
  const float* src = (c < Chalf) ? (xA + (long)b * bsA + (long)c * HWPIX)
                                 : (xB + (long)b * bsB + (long)(c - Chalf) * HWPIX);
  const float* omb = om + (long)b * 27 * HWPIX + p;
  unsigned short* dst = sbuf + (long)m * 736 + c * 9;
#pragma unroll
  for (int k = 0; k < 9; ++k) {
    float dy = omb[(long)(2 * k) * HWPIX];
    float dx = omb[(long)(2 * k + 1) * HWPIX];
    float mk = sigm(omb[(long)(18 + k) * HWPIX]);
    float py = dy + (float)(k / 3 - 1) + (float)y;
    float px = dx + (float)(k % 3 - 1) + (float)x;
    float y0f = floorf(py), x0f = floorf(px);
    float wy = py - y0f, wx = px - x0f;
    int y0 = (int)y0f, x0 = (int)x0f;
    float s = 0.0f;
#pragma unroll
    for (int cy = 0; cy < 2; ++cy) {
#pragma unroll
      for (int cx = 0; cx < 2; ++cx) {
        int yi = y0 + cy, xi = x0 + cx;
        float w = (cy ? wy : 1.0f - wy) * (cx ? wx : 1.0f - wx);
        bool valid = (yi >= 0) && (yi < IMGW) && (xi >= 0) && (xi < IMGW);
        int yc = yi < 0 ? 0 : (yi > IMGW - 1 ? IMGW - 1 : yi);
        int xc = xi < 0 ? 0 : (xi > IMGW - 1 ? IMGW - 1 : xi);
        s += src[yc * IMGW + xc] * (valid ? w : 0.0f);
      }
    }
    dst[k] = f2bf(s * mk);
  }
}

// msgs[d] += (2 - sigmoid(pA . aw + ab)) * h2   (per pixel)
__global__ void att_accum_kernel(const float* __restrict__ pA, long bsA,
                                 const float* __restrict__ aw, const float* __restrict__ ab,
                                 const float* __restrict__ h2, float* __restrict__ msg, int M)
{
  int m = blockIdx.x * blockDim.x + threadIdx.x;
  if (m >= M) return;
  int b = m >> PBITS, p = m & (HWPIX - 1);
  const float* a = pA + (long)b * bsA + p;
  float s = ab[0];
#pragma unroll
  for (int c = 0; c < 40; ++c) s += a[(long)c * HWPIX] * aw[c];
  float f = 2.0f - sigm(s);
  const float* h  = h2  + (long)b * 40 * HWPIX + p;
  float*       mm = msg + (long)b * 40 * HWPIX + p;
#pragma unroll
  for (int c = 0; c < 40; ++c) mm[(long)c * HWPIX] += f * h[(long)c * HWPIX];
}

// strided channel-block fp32 copy
__global__ void copy_ch_kernel(const float* __restrict__ src, float* __restrict__ dst,
                               long sb, long db, int C, int total)
{
  int i = blockIdx.x * blockDim.x + threadIdx.x;
  if (i >= total) return;
  int p = i & (HWPIX - 1);
  int t = i >> PBITS;
  int c = t % C, b = t / C;
  dst[(long)b * db + (long)c * HWPIX + p] = src[(long)b * sb + (long)c * HWPIX + p];
}

__global__ void zero_kernel(float* __restrict__ p, int n)
{
  int i = blockIdx.x * blockDim.x + threadIdx.x;
  if (i < n) p[i] = 0.0f;
}

// pg[b,g,p] = gin_old[g].cw[g] + cb[g] + gin_new[g].cnw[g] + cnb[g]
__global__ void cls_kernel(const float* __restrict__ bg, const float* __restrict__ nodes,
                           const float* __restrict__ newn,
                           const float* __restrict__ cw, const float* __restrict__ cb,
                           const float* __restrict__ cnw, const float* __restrict__ cnb,
                           float* __restrict__ pg, int M)
{
  int m = blockIdx.x * blockDim.x + threadIdx.x;
  if (m >= M) return;
  int b = m >> PBITS, p = m & (HWPIX - 1);
#pragma unroll
  for (int g = 0; g < 7; ++g) {
    const float* go = (g == 0) ? (bg + (long)b * 40 * HWPIX + p)
                               : (nodes + (long)b * 240 * HWPIX + (long)(g - 1) * 40 * HWPIX + p);
    const float* gn = (g == 0) ? (bg + (long)b * 40 * HWPIX + p)
                               : (newn  + (long)b * 240 * HWPIX + (long)(g - 1) * 40 * HWPIX + p);
    float s = cb[g] + cnb[g];
#pragma unroll
    for (int c = 0; c < 40; ++c)
      s += go[(long)c * HWPIX] * cw[g * 40 + c] + gn[(long)c * HWPIX] * cnw[g * 40 + c];
    pg[(long)b * 7 * HWPIX + (long)g * HWPIX + p] = s;
  }
}

// ---------------------------------------------------------------------------
extern "C" void kernel_launch(void* const* d_in, const int* in_sizes, int n_in,
                              void* d_out, int out_size, void* d_ws, size_t ws_size,
                              hipStream_t stream)
{
  (void)in_sizes; (void)n_in; (void)out_size; (void)ws_size;
  const float* xp   = (const float*)d_in[0];
  const float* xh   = (const float*)d_in[1];
  const float* xf   = (const float*)d_in[2];
  const float* p_w  = (const float*)d_in[3];
  const float* p_g  = (const float*)d_in[4];
  const float* p_b  = (const float*)d_in[5];
  const float* bg_w = (const float*)d_in[6];
  const float* bg_g = (const float*)d_in[7];
  const float* bg_b = (const float*)d_in[8];
  const float* off1w = (const float*)d_in[9];
  const float* off1b = (const float*)d_in[10];
  const float* dcn1w = (const float*)d_in[11];
  const float* bn1g  = (const float*)d_in[12];
  const float* bn1b  = (const float*)d_in[13];
  const float* off2w = (const float*)d_in[14];
  const float* off2b = (const float*)d_in[15];
  const float* dcn2w = (const float*)d_in[16];
  const float* bn2g  = (const float*)d_in[17];
  const float* bn2b  = (const float*)d_in[18];
  const float* attw  = (const float*)d_in[19];
  const float* attb  = (const float*)d_in[20];
  const float* uw1   = (const float*)d_in[21];
  const float* ug1   = (const float*)d_in[22];
  const float* ub1   = (const float*)d_in[23];
  const float* uw2   = (const float*)d_in[24];
  const float* ug2   = (const float*)d_in[25];
  const float* ub2   = (const float*)d_in[26];
  const float* clsw  = (const float*)d_in[27];
  const float* clsb  = (const float*)d_in[28];
  const float* clsnw = (const float*)d_in[29];
  const float* clsnb = (const float*)d_in[30];

  const int SRCh[10] = {0, 1, 1, 1, 2, 2, 3, 4, 4, 5};
  const int DSTh[10] = {1, 0, 2, 4, 1, 3, 2, 1, 5, 4};

  // ---------------- workspace carve-out -----------------------------------
  float* wsf = (float*)d_ws;
  size_t of = 0;
  float* nodes = wsf + of; of += (size_t)2 * 240 * HWPIX;
  float* newn  = wsf + of; of += (size_t)2 * 240 * HWPIX;
  float* bg    = wsf + of; of += (size_t)2 * 40  * HWPIX;
  float* msgs  = wsf + of; of += (size_t)6 * 2 * 40 * HWPIX;   // (6,B,40,HW)
  float* om    = wsf + of; of += (size_t)2 * 27 * HWPIX;
  float* h1    = wsf + of; of += (size_t)2 * 80 * HWPIX;
  float* h2    = wsf + of; of += (size_t)2 * 40 * HWPIX;
  // bf16 region (16B aligned: float offset is a multiple of 4)
  unsigned short* wss = (unsigned short*)(wsf + of);
  size_t os = 0;
  unsigned short* cat   = wss + os; os += (size_t)MTOT * 768;  // aliased with sbuf
  unsigned short* sbuf  = cat;                                 // (M,736) after bg stage
  unsigned short* updin = wss + os; os += (size_t)MTOT * 96;
  unsigned short* u1    = wss + os; os += (size_t)MTOT * 32;
  // N-padded bf16 weights
  unsigned short* pwb    = wss + os; os += (size_t)256 * 256;   // 240->256
  unsigned short* bgwb   = wss + os; os += (size_t)64  * 768;   // 40->64
  unsigned short* off1wb = wss + os; os += (size_t)10 * 32 * 736;  // 27->32
  unsigned short* dcn1wb = wss + os; os += (size_t)10 * 96 * 736;  // 80->96
  unsigned short* off2wb = wss + os; os += (size_t)10 * 32 * 736;
  unsigned short* dcn2wb = wss + os; os += (size_t)10 * 64 * 736;  // 40->64
  unsigned short* uw1b   = wss + os; os += (size_t)6 * 32 * 96;
  unsigned short* uw2b   = wss + os; os += (size_t)6 * 64 * 32;    // 40->64

  float* out    = (float*)d_out;                 // xp_infer (B,536,HW)
  float* pg_out = out + (size_t)2 * 536 * HWPIX;

  auto WCONV = [&](const float* s, unsigned short* d, int NGRP, int NPAD,
                   int KR, int KP, int groups) {
    int total = groups * NPAD * KP;
    wconv_kernel<<<(total + 255) / 256, 256, 0, stream>>>(s, d, NGRP, NPAD, KR, KP, total);
  };
  auto PACK = [&](const float* s, long sb, unsigned short* d, long dl, int c0, int C) {
    int total = MTOT * C;
    pack_rows_kernel<<<(total + 255) / 256, 256, 0, stream>>>(s, sb, d, dl, c0, C, total);
  };
  auto COPY = [&](const float* s, float* d, long sb, long db, int C) {
    int total = 2 * C * HWPIX;
    copy_ch_kernel<<<(total + 255) / 256, 256, 0, stream>>>(s, d, sb, db, C, total);
  };

  // ---- one-time weight conversion to N/K padded bf16 ----------------------
  WCONV(p_w,   pwb,    240, 256, 256, 256, 1);
  WCONV(bg_w,  bgwb,   40,  64,  768, 768, 1);
  WCONV(off1w, off1wb, 27,  32,  720, 736, 10);
  WCONV(dcn1w, dcn1wb, 80,  96,  720, 736, 10);
  WCONV(off2w, off2wb, 27,  32,  720, 736, 10);
  WCONV(dcn2w, dcn2wb, 40,  64,  720, 736, 10);
  WCONV(uw1,   uw1b,   32,  32,  80,  96,  6);
  WCONV(uw2,   uw2b,   40,  64,  32,  32,  6);

  // ---- cat = [xp|xh|xf] as row-major bf16 (M,768) -------------------------
  PACK(xp, 256L * HWPIX, cat, 768, 0,   256);
  PACK(xh, 256L * HWPIX, cat, 768, 256, 256);
  PACK(xf, 256L * HWPIX, cat, 768, 512, 256);

  dim3 gN256(MTOT / 64, 8), gN64(MTOT / 64, 2), gN32(MTOT / 64, 1),
       gN96(MTOT / 64, 3);

  // ---- node projection (K=256 = first 256 cols of cat) & bg (K=768) -------
  gemm_bf16_kernel<256><<<gN256, 128, 0, stream>>>(cat, 768, pwb, 256, p_g, p_b,
      nodes, nullptr, 240L * HWPIX, 240, 1);
  gemm_bf16_kernel<768><<<gN64, 128, 0, stream>>>(cat, 768, bgwb, 768, bg_g, bg_b,
      bg, nullptr, 40L * HWPIX, 40, 1);

  // ---- repurpose cat region as sbuf (M,736): zero its pad columns ---------
  {
    int total = MTOT * 16;
    zero_cols_kernel<<<(total + 255) / 256, 256, 0, stream>>>(sbuf, 736, 720, 16, total);
    zero_cols_kernel<<<(total + 255) / 256, 256, 0, stream>>>(updin, 96, 80, 16, total);
    int tz = 6 * 2 * 40 * HWPIX;
    zero_kernel<<<(tz + 255) / 256, 256, 0, stream>>>(msgs, tz);
  }

  // ---- edge loop ----------------------------------------------------------
  const int MC = MTOT * 80;
  const int ebl = (MC + 255) / 256;
  for (int e = 0; e < 10; ++e) {
    const float* pA = nodes + (long)SRCh[e] * 40 * HWPIX;
    const float* pB = nodes + (long)DSTh[e] * 40 * HWPIX;
    const long nbs = 240L * HWPIX;

    im2col_kernel<<<ebl, 256, 0, stream>>>(pA, pB, nbs, nbs, sbuf, 80, 40, MTOT);
    gemm_bf16_kernel<736><<<gN32, 128, 0, stream>>>(sbuf, 736, off1wb + (long)e * 32 * 736,
        736, nullptr, off1b + e * 27, om, nullptr, 27L * HWPIX, 27, 0);

    dcn_sample_kernel<<<ebl, 256, 0, stream>>>(pA, pB, nbs, nbs, om, sbuf, 80, 40, MTOT);
    gemm_bf16_kernel<736><<<gN96, 128, 0, stream>>>(sbuf, 736, dcn1wb + (long)e * 96 * 736,
        736, bn1g + e * 80, bn1b + e * 80, h1, nullptr, 80L * HWPIX, 80, 1);

    im2col_kernel<<<ebl, 256, 0, stream>>>(h1, h1 + 40 * HWPIX, 80L * HWPIX, 80L * HWPIX,
                                           sbuf, 80, 40, MTOT);
    gemm_bf16_kernel<736><<<gN32, 128, 0, stream>>>(sbuf, 736, off2wb + (long)e * 32 * 736,
        736, nullptr, off2b + e * 27, om, nullptr, 27L * HWPIX, 27, 0);

    dcn_sample_kernel<<<ebl, 256, 0, stream>>>(h1, h1 + 40 * HWPIX, 80L * HWPIX, 80L * HWPIX,
                                               om, sbuf, 80, 40, MTOT);
    gemm_bf16_kernel<736><<<gN64, 128, 0, stream>>>(sbuf, 736, dcn2wb + (long)e * 64 * 736,
        736, bn2g + e * 40, bn2b + e * 40, h2, nullptr, 40L * HWPIX, 40, 1);

    att_accum_kernel<<<(MTOT + 255) / 256, 256, 0, stream>>>(
        pA, nbs, attw + e * 40, attb + e,
        h2, msgs + (long)DSTh[e] * 2 * 40 * HWPIX, MTOT);
  }

  // ---- node update: pack [node|msg] -> (M,96) bf16, two WMMA GEMMs --------
  for (int n = 0; n < 6; ++n) {
    PACK(nodes + (long)n * 40 * HWPIX,     240L * HWPIX, updin, 96, 0,  40);
    PACK(msgs + (long)n * 2 * 40 * HWPIX,  40L * HWPIX,  updin, 96, 40, 40);
    gemm_bf16_kernel<96><<<gN32, 128, 0, stream>>>(updin, 96, uw1b + n * 32 * 96, 96,
        ug1 + n * 32, ub1 + n * 32, nullptr, u1, 32, 32, 1);
    gemm_bf16_kernel<32><<<gN64, 128, 0, stream>>>(u1, 32, uw2b + n * 64 * 32, 32,
        ug2 + n * 40, ub2 + n * 40, newn + (long)n * 40 * HWPIX, nullptr,
        240L * HWPIX, 40, 1);
  }

  // ---- assemble xp_infer = [xp | bg | new_nodes] --------------------------
  COPY(xp,   out,                256L * HWPIX, 536L * HWPIX, 256);
  COPY(bg,   out + 256 * HWPIX,  40L * HWPIX,  536L * HWPIX, 40);
  COPY(newn, out + 296 * HWPIX,  240L * HWPIX, 536L * HWPIX, 240);

  // ---- classifier heads -> pg (B,7,HW) ------------------------------------
  cls_kernel<<<(MTOT + 255) / 256, 256, 0, stream>>>(
      bg, nodes, newn, clsw, clsb, clsnw, clsnb, pg_out, MTOT);
}
